// HierarchicalDecoder_72292889527048
// MI455X (gfx1250) — matrix operand order repaired and checked by reading, not compiled, not run on hardware
//
#include <hip/hip_runtime.h>
#include <math.h>

typedef __attribute__((ext_vector_type(16))) _Float16 v16h;
typedef __attribute__((ext_vector_type(8)))  float    v8f;

namespace {
constexpr int BATCH = 256;
constexpr int CHID  = 1024;   // conductor hidden
constexpr int COE   = 512;    // conductor embedding
constexpr int DHID  = 1024;   // decoder hidden
constexpr int NOTE  = 388;    // note vocab
constexpr int NOTEP = 400;    // padded vocab (25 tiles of 16)
constexpr int LATD  = 512;    // latent
constexpr int NSEGc = 16;
constexpr int SLENc = 16;
constexpr int TSTEP = NSEGc * SLENc;   // 256
constexpr int XK    = 1024;   // padded decoder layer-0 input (388+512=900 -> 1024)
}

// ---------------------------------------------------------------------------
// WMMA helpers: 16x16x32 f16 -> f32 accumulate (CDNA5 wave32 layouts)
// ---------------------------------------------------------------------------
__device__ __forceinline__ v8f wmma16(v16h a, v16h b, v8f c) {
  return __builtin_amdgcn_wmma_f32_16x16x32_f16(
      /*neg_a=*/false, a, /*neg_b=*/false, b,
      /*c_mod=*/(short)0, c, /*reuse_a=*/false, /*reuse_b=*/false);
}

// A tile 16x32 (MxK), row-major source X[ld = ldx]:
//  lane<16: row=lane,    halves0..15 = K[k0..k0+15]
//  lane>=16: row=lane-16, halves0..15 = K[k0+16..k0+31]
__device__ __forceinline__ v16h load_a_tile(const _Float16* __restrict__ X,
                                            int ldx, int m0, int k0, int lane) {
  int row = m0 + (lane & 15);
  int hi  = lane >> 4;
  const _Float16* p = X + (size_t)row * ldx + k0 + 16 * hi;
  union { v16h v; uint4 u[2]; } a;
  a.u[0] = *reinterpret_cast<const uint4*>(p);
  a.u[1] = *reinterpret_cast<const uint4*>(p + 8);
  return a.v;
}

// B tile 32x16 (KxN) built from row-major W[N,K] (B col n == W row n):
//  lane<16: col=lane,    halves0..15 = K[k0..k0+15]
//  lane>=16: col=lane-16, halves0..15 = K[k0+16..k0+31]
__device__ __forceinline__ v16h load_b_tile(const _Float16* __restrict__ W,
                                            int ldw, int n0, int k0, int lane) {
  int col = n0 + (lane & 15);
  int hi  = lane >> 4;
  const _Float16* p = W + (size_t)col * ldw + k0 + 16 * hi;
  union { v16h v; uint4 u[2]; } b;
  b.u[0] = *reinterpret_cast<const uint4*>(p);
  b.u[1] = *reinterpret_cast<const uint4*>(p + 8);
  return b.v;
}

__device__ __forceinline__ float sigm(float x) { return 1.0f / (1.0f + expf(-x)); }

// ---- CDNA5 async global->LDS staging -------------------------------------
__device__ __forceinline__ void async_ld_b128(unsigned lds_off, const _Float16* g) {
  asm volatile("global_load_async_to_lds_b128 %0, %1, off"
               :: "v"(lds_off), "v"((unsigned long long)(size_t)g)
               : "memory");
}
__device__ __forceinline__ void wait_async_le1() {
  asm volatile("s_wait_asynccnt 0x1" ::: "memory");
}
__device__ __forceinline__ void wait_async_0() {
  asm volatile("s_wait_asynccnt 0x0" ::: "memory");
}

// ---------------------------------------------------------------------------
// Fused LSTM cell: g = X@Wih^T + Hin@Whh^T + b ; c' = sig(f)c + sig(i)tanh(g);
// h' = sig(o)tanh(c').
// Block = 256 threads (8 waves); each wave owns one 16(batch)x16(hid) tile
// with 4 gate accumulators. The 4 gate B-tiles (4KB per K-step) are staged
// once per block into double-buffered LDS via global_load_async_to_lds_b128
// (ASYNCcnt-tracked) and shared by all 8 waves. K loop is split into two
// branch-free phases (Wih/X then Whh/Hin) with peeled boundary + epilogue.
// grid = (H/16, BATCH/128)
// ---------------------------------------------------------------------------
__global__ __launch_bounds__(256) void lstm_gates_kernel(
    const _Float16* __restrict__ X, int Kx, int ldx,
    const _Float16* __restrict__ Wih,          // [4H, Kx] row-major f16
    const _Float16* __restrict__ Hin,          // [BATCH, H]
    const _Float16* __restrict__ Whh,          // [4H, H]
    const float*    __restrict__ bias,         // [4H]
    float*          __restrict__ C,            // [BATCH, H] fp32, in-place
    _Float16*       __restrict__ Hout,         // [BATCH, H]
    int H) {
  constexpr int TROW = 40;          // halves per tile row (padded: conflict-free)
  constexpr int TSZ  = 16 * TROW;   // 640 halves per gate tile
  constexpr int BUFH = 4 * TSZ;     // 2560 halves per buffer
  __shared__ __align__(16) _Float16 smem[2][BUFH];

  const int tid  = threadIdx.x;
  const int lane = tid & 31;
  const int wave = tid >> 5;
  const int m0 = blockIdx.y * 128 + wave * 16;
  const int n0 = blockIdx.x * 16;

  // loader mapping: one b128 (8 halves) per thread per K-step
  const int lg = tid >> 6;          // gate 0..3
  const int lr = (tid >> 2) & 15;   // row within tile (weight row)
  const int lc = tid & 3;           // 8-half chunk 0..3
  const unsigned ldsbase =
      (unsigned)(unsigned long long)(const void*)&smem[0][0];
  const unsigned ldstw = ldsbase + 2u * (lg * TSZ + lr * TROW + lc * 8);

  const int nkx = Kx >> 5;
  const int nkh = H >> 5;
  const int nk  = nkx + nkh;

  // per-thread weight source pointers
  const _Float16* gih = Wih + (size_t)(lg * H + n0 + lr) * Kx + lc * 8;
  const _Float16* ghh = Whh + (size_t)(lg * H + n0 + lr) * H  + lc * 8;

  // per-lane A-tile pointers
  const int arow = m0 + (lane & 15);
  const int acol = (lane >> 4) * 16;
  const _Float16* axp = X   + (size_t)arow * ldx + acol;
  const _Float16* ahp = Hin + (size_t)arow * H   + acol;

  const _Float16* sbase = &smem[0][(lane & 15) * TROW + (lane >> 4) * 16];

  v8f ai = {}, af = {}, ag = {}, ao = {};

  auto issue = [&](int idx, const _Float16* g) {
    async_ld_b128(ldstw + (unsigned)(idx & 1) * (unsigned)(2 * BUFH), g);
  };
  auto consume = [&](int idx, const _Float16* apt) {
    union { v16h v; uint4 u[2]; } a, b;
    a.u[0] = *(const uint4*)apt;
    a.u[1] = *(const uint4*)(apt + 8);
    const _Float16* sb = sbase + (idx & 1) * BUFH;
    b.u[0] = *(const uint4*)(sb + 0 * TSZ); b.u[1] = *(const uint4*)(sb + 0 * TSZ + 8);
    ai = wmma16(a.v, b.v, ai);
    b.u[0] = *(const uint4*)(sb + 1 * TSZ); b.u[1] = *(const uint4*)(sb + 1 * TSZ + 8);
    af = wmma16(a.v, b.v, af);
    b.u[0] = *(const uint4*)(sb + 2 * TSZ); b.u[1] = *(const uint4*)(sb + 2 * TSZ + 8);
    ag = wmma16(a.v, b.v, ag);
    b.u[0] = *(const uint4*)(sb + 3 * TSZ); b.u[1] = *(const uint4*)(sb + 3 * TSZ + 8);
    ao = wmma16(a.v, b.v, ao);
  };

  issue(0, gih);
  // phase 1: issue Wih tiles, consume X tiles (branch-free body)
  for (int i = 0; i < nkx - 1; ++i) {
    issue(i + 1, gih + (size_t)(i + 1) * 32);
    wait_async_le1();
    __syncthreads();
    consume(i, axp + (size_t)i * 32);
    __syncthreads();
  }
  // boundary: issue first Whh tile, consume last X tile
  issue(nkx, ghh);
  wait_async_le1();
  __syncthreads();
  consume(nkx - 1, axp + (size_t)(nkx - 1) * 32);
  __syncthreads();
  // phase 2: issue Whh tiles, consume Hin tiles (branch-free body)
  for (int i = nkx; i < nk - 1; ++i) {
    issue(i + 1, ghh + (size_t)(i + 1 - nkx) * 32);
    wait_async_le1();
    __syncthreads();
    consume(i, ahp + (size_t)(i - nkx) * 32);
    __syncthreads();
  }
  // epilogue
  wait_async_0();
  __syncthreads();
  consume(nk - 1, ahp + (size_t)(nkh - 1) * 32);

  const int j  = n0 + (lane & 15);
  const int hi = lane >> 4;
  const float bi = bias[0 * H + j];
  const float bf = bias[1 * H + j];
  const float bg = bias[2 * H + j];
  const float bo = bias[3 * H + j];

#pragma unroll
  for (int r = 0; r < 8; ++r) {
    const int m = m0 + r + 8 * hi;
    const size_t idx = (size_t)m * H + j;
    const float gi = sigm(ai[r] + bi);
    const float gf = sigm(af[r] + bf);
    const float gg = tanhf(ag[r] + bg);
    const float go = sigm(ao[r] + bo);
    const float cn = gf * C[idx] + gi * gg;
    C[idx] = cn;
    Hout[idx] = (_Float16)(go * tanhf(cn));
  }
}

// ---------------------------------------------------------------------------
// Generic WMMA GEMM: Out[m, col_off+n] = act(X@W^T + bias). ACT: 0=id, 1=tanh
// grid = (N/16, BATCH/64), block 128
// ---------------------------------------------------------------------------
template <int ACT, typename OutT>
__global__ __launch_bounds__(128) void gemm_kernel(
    const _Float16* __restrict__ X, int ldx,
    const _Float16* __restrict__ W, int ldw,
    const float* __restrict__ bias, int nbias,
    OutT* __restrict__ Out, int ldo, int col_off, int K) {
  const int lane = threadIdx.x & 31;
  const int wave = threadIdx.x >> 5;
  const int m0 = blockIdx.y * 64 + wave * 16;
  const int n0 = blockIdx.x * 16;

  v8f acc = {};
  for (int k0 = 0; k0 < K; k0 += 32) {
    v16h a = load_a_tile(X, ldx, m0, k0, lane);
    v16h b = load_b_tile(W, ldw, n0, k0, lane);
    acc = wmma16(a, b, acc);
  }

  const int j  = n0 + (lane & 15);
  const int hi = lane >> 4;
  const float bv = (j < nbias) ? bias[j] : 0.0f;
#pragma unroll
  for (int r = 0; r < 8; ++r) {
    float v = acc[r] + bv;
    if (ACT == 1) v = tanhf(v);
    Out[(size_t)(m0 + r + 8 * hi) * ldo + col_off + j] = (OutT)v;
  }
}

// ---------------------------------------------------------------------------
// fp32 -> f16 conversion with row/col zero padding
// ---------------------------------------------------------------------------
__global__ void cvt_pad_kernel(const float* __restrict__ src, _Float16* __restrict__ dst,
                               int rsrc, int ksrc, int kdst, int total) {
  int i = blockIdx.x * blockDim.x + threadIdx.x;
  if (i >= total) return;
  int r = i / kdst;
  int k = i - r * kdst;
  float v = (r < rsrc && k < ksrc) ? src[(size_t)r * ksrc + k] : 0.0f;
  dst[i] = (_Float16)v;
}

// ---------------------------------------------------------------------------
// Row softmax over 388 logits; writes fp32 output and f16 prev-note back into
// the decoder input buffer. One wave per batch row. grid = BATCH/4, block 128.
// ---------------------------------------------------------------------------
__global__ __launch_bounds__(128) void softmax_kernel(
    const float* __restrict__ logits, int ldl,
    float* __restrict__ out, int t,
    _Float16* __restrict__ xprev, int ldxp) {
  const int lane = threadIdx.x & 31;
  const int wave = threadIdx.x >> 5;
  const int row  = blockIdx.x * 4 + wave;
  const float* lp = logits + (size_t)row * ldl;

  float mx = -1e30f;
  for (int n = lane; n < NOTE; n += 32) mx = fmaxf(mx, lp[n]);
  for (int off = 16; off > 0; off >>= 1) mx = fmaxf(mx, __shfl_xor(mx, off, 32));

  float s = 0.0f;
  for (int n = lane; n < NOTE; n += 32) s += expf(lp[n] - mx);
  for (int off = 16; off > 0; off >>= 1) s += __shfl_xor(s, off, 32);
  const float inv = 1.0f / s;

  float* op = out + (size_t)row * TSTEP * NOTE + (size_t)t * NOTE;
  _Float16* xp = xprev + (size_t)row * ldxp;
  for (int n = lane; n < NOTE; n += 32) {
    float p = expf(lp[n] - mx) * inv;
    op[n] = p;
    xp[n] = (_Float16)p;
  }
}

// ---------------------------------------------------------------------------
// Host driver
// ---------------------------------------------------------------------------
extern "C" void kernel_launch(void* const* d_in, const int* in_sizes, int n_in,
                              void* d_out, int out_size, void* d_ws, size_t ws_size,
                              hipStream_t stream) {
  (void)in_sizes; (void)n_in; (void)out_size; (void)ws_size;

  const float* z      = (const float*)d_in[0];
  const float* projW  = (const float*)d_in[1];
  const float* projb  = (const float*)d_in[2];
  const float* cWih0  = (const float*)d_in[3];
  const float* cWhh0  = (const float*)d_in[4];
  const float* cb0    = (const float*)d_in[5];
  const float* cWih1  = (const float*)d_in[6];
  const float* cWhh1  = (const float*)d_in[7];
  const float* cb1    = (const float*)d_in[8];
  const float* cfcW1  = (const float*)d_in[9];
  const float* cfcb1  = (const float*)d_in[10];
  const float* cfcW2  = (const float*)d_in[11];
  const float* cfcb2  = (const float*)d_in[12];
  const float* dWih0  = (const float*)d_in[13];
  const float* dWhh0  = (const float*)d_in[14];
  const float* db0    = (const float*)d_in[15];
  const float* dWih1  = (const float*)d_in[16];
  const float* dWhh1  = (const float*)d_in[17];
  const float* db1    = (const float*)d_in[18];
  const float* dfcW   = (const float*)d_in[19];
  const float* dfcb   = (const float*)d_in[20];

  // ---- workspace layout (256B aligned) ----
  char* ws = (char*)d_ws;
  size_t off = 0;
  auto a16 = [&](size_t elems) -> _Float16* {
    _Float16* p = (_Float16*)(ws + off);
    off += ((elems * 2 + 255) & ~(size_t)255);
    return p;
  };
  auto a32 = [&](size_t elems) -> float* {
    float* p = (float*)(ws + off);
    off += ((elems * 4 + 255) & ~(size_t)255);
    return p;
  };

  // f16 weights (converted once per call; ~68 MB, L2 resident thereafter)
  _Float16* wProj  = a16((size_t)CHID * LATD);
  _Float16* wCWih0 = a16((size_t)4 * CHID * CHID);
  _Float16* wCWhh0 = a16((size_t)4 * CHID * CHID);
  _Float16* wCWih1 = a16((size_t)4 * CHID * CHID);
  _Float16* wCWhh1 = a16((size_t)4 * CHID * CHID);
  _Float16* wCfc1  = a16((size_t)COE * CHID);
  _Float16* wCfc2  = a16((size_t)COE * COE);
  _Float16* wDWih0 = a16((size_t)4 * DHID * XK);      // K padded 900 -> 1024
  _Float16* wDWhh0 = a16((size_t)4 * DHID * DHID);
  _Float16* wDWih1 = a16((size_t)4 * DHID * DHID);
  _Float16* wDWhh1 = a16((size_t)4 * DHID * DHID);
  _Float16* wDfc   = a16((size_t)NOTEP * DHID);       // rows padded 388 -> 400

  // f16 activations
  _Float16* z16  = a16((size_t)BATCH * LATD);
  _Float16* zc   = a16((size_t)BATCH * CHID);
  _Float16* ch0a = a16((size_t)BATCH * CHID);
  _Float16* ch0b = a16((size_t)BATCH * CHID);
  _Float16* ch1a = a16((size_t)BATCH * CHID);
  _Float16* ch1b = a16((size_t)BATCH * CHID);
  _Float16* dh0a = a16((size_t)BATCH * DHID);
  _Float16* dh0b = a16((size_t)BATCH * DHID);
  _Float16* dh1a = a16((size_t)BATCH * DHID);
  _Float16* dh1b = a16((size_t)BATCH * DHID);
  _Float16* xbuf = a16((size_t)BATCH * XK);           // [prev(388) | emb(512) | pad]
  _Float16* cfct = a16((size_t)BATCH * COE);

  // fp32 state / scratch
  float* cc0    = a32((size_t)BATCH * CHID);
  float* cc1    = a32((size_t)BATCH * CHID);
  float* dc0    = a32((size_t)BATCH * DHID);
  float* dc1    = a32((size_t)BATCH * DHID);
  float* logits = a32((size_t)BATCH * NOTEP);

  // ---- convert weights to f16 (with padding) ----
  auto cvt = [&](const float* src, _Float16* dst, int rsrc, int rdst, int ksrc, int kdst) {
    int total = rdst * kdst;
    cvt_pad_kernel<<<(total + 255) / 256, 256, 0, stream>>>(src, dst, rsrc, ksrc, kdst, total);
  };
  cvt(projW, wProj,  CHID,     CHID,     LATD, LATD);
  cvt(cWih0, wCWih0, 4 * CHID, 4 * CHID, CHID, CHID);
  cvt(cWhh0, wCWhh0, 4 * CHID, 4 * CHID, CHID, CHID);
  cvt(cWih1, wCWih1, 4 * CHID, 4 * CHID, CHID, CHID);
  cvt(cWhh1, wCWhh1, 4 * CHID, 4 * CHID, CHID, CHID);
  cvt(cfcW1, wCfc1,  COE,      COE,      CHID, CHID);
  cvt(cfcW2, wCfc2,  COE,      COE,      COE,  COE);
  cvt(dWih0, wDWih0, 4 * DHID, 4 * DHID, NOTE + COE, XK);   // 900 -> 1024
  cvt(dWhh0, wDWhh0, 4 * DHID, 4 * DHID, DHID, DHID);
  cvt(dWih1, wDWih1, 4 * DHID, 4 * DHID, DHID, DHID);
  cvt(dWhh1, wDWhh1, 4 * DHID, 4 * DHID, DHID, DHID);
  cvt(dfcW,  wDfc,   NOTE,     NOTEP,    DHID, DHID);       // 388 -> 400 rows
  cvt(z,     z16,    BATCH,    BATCH,    LATD, LATD);

  // ---- zero initial states (0x00 == 0.0f == f16 zero) ----
  hipMemsetAsync(ch0a, 0, (size_t)BATCH * CHID * 2, stream);
  hipMemsetAsync(ch1a, 0, (size_t)BATCH * CHID * 2, stream);
  hipMemsetAsync(dh0a, 0, (size_t)BATCH * DHID * 2, stream);
  hipMemsetAsync(dh1a, 0, (size_t)BATCH * DHID * 2, stream);
  hipMemsetAsync(xbuf, 0, (size_t)BATCH * XK * 2, stream);
  hipMemsetAsync(cc0, 0, (size_t)BATCH * CHID * 4, stream);
  hipMemsetAsync(cc1, 0, (size_t)BATCH * CHID * 4, stream);
  hipMemsetAsync(dc0, 0, (size_t)BATCH * DHID * 4, stream);
  hipMemsetAsync(dc1, 0, (size_t)BATCH * DHID * 4, stream);

  // ---- launch helpers ----
  auto lstm = [&](const _Float16* X, int Kx, int ldx, const _Float16* Wih,
                  const _Float16* Hin, const _Float16* Whh, const float* bias,
                  float* C, _Float16* Hout, int H) {
    dim3 g(H / 16, BATCH / 128);
    lstm_gates_kernel<<<g, 256, 0, stream>>>(X, Kx, ldx, Wih, Hin, Whh, bias, C, Hout, H);
  };

  // ---- conductor input: zc = tanh(z @ projW^T + projb), f16 [256,1024] ----
  {
    dim3 g(CHID / 16, BATCH / 64);
    gemm_kernel<1, _Float16><<<g, 128, 0, stream>>>(
        z16, LATD, wProj, LATD, projb, CHID, zc, CHID, 0, LATD);
  }

  _Float16 *c0c = ch0a, *c0n = ch0b, *c1c = ch1a, *c1n = ch1b;
  _Float16 *d0c = dh0a, *d0n = dh0b, *d1c = dh1a, *d1n = dh1b;
  _Float16* tmp;

  for (int seg = 0; seg < NSEGc; ++seg) {
    // conductor step (2-layer LSTM)
    lstm(zc, CHID, CHID, wCWih0, c0c, wCWhh0, cb0, cc0, c0n, CHID);
    tmp = c0c; c0c = c0n; c0n = tmp;
    lstm(c0c, CHID, CHID, wCWih1, c1c, wCWhh1, cb1, cc1, c1n, CHID);
    tmp = c1c; c1c = c1n; c1n = tmp;

    // emb = tanh((ch1 @ W1^T + b1) @ W2^T + b2) -> xbuf[:, 388:900]
    {
      dim3 g1(COE / 16, BATCH / 64);
      gemm_kernel<0, _Float16><<<g1, 128, 0, stream>>>(
          c1c, CHID, wCfc1, CHID, cfcb1, COE, cfct, COE, 0, CHID);
      gemm_kernel<1, _Float16><<<g1, 128, 0, stream>>>(
          cfct, COE, wCfc2, COE, cfcb2, COE, xbuf, XK, NOTE, COE);
    }

    for (int s = 0; s < SLENc; ++s) {
      // decoder step (2-layer LSTM on [prev | emb])
      lstm(xbuf, XK, XK, wDWih0, d0c, wDWhh0, db0, dc0, d0n, DHID);
      tmp = d0c; d0c = d0n; d0n = tmp;
      lstm(d0c, DHID, DHID, wDWih1, d1c, wDWhh1, db1, dc1, d1n, DHID);
      tmp = d1c; d1c = d1n; d1n = tmp;

      // logits = dh1 @ dfcW^T + dfcb  (N padded to 400)
      {
        dim3 g(NOTEP / 16, BATCH / 64);
        gemm_kernel<0, float><<<g, 128, 0, stream>>>(
            d1c, DHID, wDfc, DHID, dfcb, NOTE, logits, NOTEP, 0, DHID);
      }
      // softmax -> d_out[:, t, :] and prev-note f16 -> xbuf[:, 0:388]
      softmax_kernel<<<BATCH / 4, 128, 0, stream>>>(
          logits, NOTEP, (float*)d_out, seg * SLENc + s, xbuf, XK);
    }
  }
}
